// FrequencyAwareAttention_1279900254224
// MI455X (gfx1250) — compile-verified, matrix operand order checked by reading, not run on hardware
//
#include <hip/hip_runtime.h>
#include <hip/hip_bf16.h>
#include <math.h>

typedef __bf16 bf16_t;
typedef __attribute__((ext_vector_type(8)))  __bf16 v8bf;
typedef __attribute__((ext_vector_type(16))) __bf16 v16bf;
typedef __attribute__((ext_vector_type(8)))  float  v8f;

#define D_MODEL 512
#define LSEQ    4096
#define NB      2
#define NH      8
#define DKH     64
#define HALFL   2048
#define CHK     256   // key chunk for attention softmax

#define WMMA_BF16(a,b,c) \
  __builtin_amdgcn_wmma_f32_16x16x32_bf16(false,(a),false,(b),(short)0,(c),false,false)

// ---------------------------------------------------------------------------
// Fragment loader for 16x32 bf16 tiles (A row-major, or B stored N-major).
// CDNA5 16-bit A-matrix layout: lane L -> row (L&15); K runs
// [base..base+7] and [16+base..16+base+7], base = (L>>4)*8.
// Both runs are contiguous -> two b128 loads.
// ---------------------------------------------------------------------------
__device__ inline v16bf load_frag(const bf16_t* tile, int stride) {
  int lane = threadIdx.x & 31;
  int r    = lane & 15;
  int kb   = (lane >> 4) << 3;
  const bf16_t* p = tile + (long)r * stride + kb;
  v8bf lo = *(const v8bf*)(p);
  v8bf hi = *(const v8bf*)(p + 16);
  v16bf f;
#pragma unroll
  for (int i = 0; i < 8; ++i) { f[i] = lo[i]; f[i + 8] = hi[i]; }
  return f;
}

// ---------------------------------------------------------------------------
// Generic WMMA GEMM with A-fragment reuse: each wave owns a 16x64 output
// strip (4 accumulators), so one A fragment feeds 4 WMMAs per k-step.
// out[z, rowBase+m, ncolOff+col] = A(MxK) * Wt(N-major KxN) + bias (+GELU).
// grid = (N/256, M/16, NB), block = 128 (4 waves).
// ---------------------------------------------------------------------------
__global__ __launch_bounds__(128)
void gemm_bf16(const bf16_t* __restrict__ A, long asb, int lda,
               const bf16_t* __restrict__ Wt, int K,
               const float* __restrict__ bias,
               float* __restrict__ outF, bf16_t* __restrict__ outB,
               long osb, int ldo, int ncolOff, int gelu)
{
  int z       = blockIdx.z;
  int w       = threadIdx.x >> 5;
  int lane    = threadIdx.x & 31;
  int colBase = blockIdx.x * 256 + w * 64;   // wave: cols [colBase, colBase+64)
  int rowBase = blockIdx.y * 16;

  const bf16_t* Ab = A + z * asb + (long)rowBase * lda;
  const bf16_t* Bb = Wt + (long)colBase * K;

  v8f acc[4] = {};
  for (int k = 0; k < K; k += 32) {
    // prefetch next k-step's cachelines (global_prefetch_b8)
    __builtin_prefetch(Ab + k + 32, 0, 1);
    __builtin_prefetch(Bb + k + 32, 0, 1);
    v16bf af = load_frag(Ab + k, lda);
#pragma unroll
    for (int t = 0; t < 4; ++t) {
      v16bf bfr = load_frag(Bb + (long)(t * 16) * K + k, K);
      acc[t] = WMMA_BF16(af, bfr, acc[t]);
    }
  }

  int n  = lane & 15;
  int rg = (lane >> 4) << 3;
#pragma unroll
  for (int t = 0; t < 4; ++t) {
    int col = colBase + t * 16 + n;
    float bb = bias[col];
#pragma unroll
    for (int j = 0; j < 8; ++j) {
      int r = rowBase + rg + j;
      float v = acc[t][j] + bb;
      if (gelu) v = 0.5f * v * (1.0f + erff(v * 0.70710678118f));
      long idx = z * osb + (long)r * ldo + ncolOff + col;
      if (outF) outF[idx] = v;
      if (outB) outB[idx] = (bf16_t)v;
    }
  }
}

// ---------------------------------------------------------------------------
// Score chunk: S[16][CHK] = scale * Q(16xdk) * K^T(dk x CHK), optional band.
// ---------------------------------------------------------------------------
__device__ inline void score_chunk(float S[16][CHK], const bf16_t* Kb,
                                   int c0, int qbase, v16bf aq0, v16bf aq1,
                                   int band, int win)
{
  int w    = threadIdx.x >> 5;
  int lane = threadIdx.x & 31;
  for (int nt = w; nt < CHK / 16; nt += 4) {
    int keyTile = c0 + nt * 16;
    v8f s = {};
    v16bf b0 = load_frag(Kb + (long)keyTile * D_MODEL, D_MODEL);
    s = WMMA_BF16(aq0, b0, s);
    v16bf b1 = load_frag(Kb + (long)keyTile * D_MODEL + 32, D_MODEL);
    s = WMMA_BF16(aq1, b1, s);
    int n  = lane & 15;
    int key = keyTile + n;
    int rg = (lane >> 4) << 3;
#pragma unroll
    for (int j = 0; j < 8; ++j) {
      int m = rg + j;
      float v = s[j] * 0.125f;           // 1/sqrt(64)
      if (band) {
        int d = (qbase + m) - key; if (d < 0) d = -d;
        if (d > win) v = -1.0e9f;
      }
      S[m][nt * 16 + n] = v;
    }
  }
}

// ---------------------------------------------------------------------------
// Attention for one (batch, head, 16-query tile).  Two-sweep softmax over
// key chunks of CHK; sweep2 accumulates O = P*V with WMMA.  V is dk-major.
// grid = (Lq/16, NH, NB), block = 128 (4 waves; wave w owns dims 16w..16w+15)
// ---------------------------------------------------------------------------
__global__ __launch_bounds__(128)
void attn_kernel(const bf16_t* __restrict__ Q, long qsb,
                 const bf16_t* __restrict__ Kc, long ksb,
                 const bf16_t* __restrict__ Vt, int Lk,
                 bf16_t* __restrict__ O, long osb,
                 int band, int win)
{
  __shared__ float  S[16][CHK];
  __shared__ __align__(16) bf16_t P[16][CHK];
  __shared__ float rmax[16];
  __shared__ float rsum[16];

  int z = blockIdx.z, h = blockIdx.y;
  int qbase = blockIdx.x * 16;
  int w    = threadIdx.x >> 5;
  int lane = threadIdx.x & 31;
  int tid  = threadIdx.x;

  const bf16_t* Qb = Q + z * qsb + (long)qbase * D_MODEL + h * DKH;
  const bf16_t* Kb = Kc + z * ksb + h * DKH;
  const bf16_t* Vb = Vt + (long)(z * NH + h) * DKH * Lk;

  v16bf aq0 = load_frag(Qb, D_MODEL);
  v16bf aq1 = load_frag(Qb + 32, D_MODEL);

  if (tid < 16) { rmax[tid] = -3.0e38f; rsum[tid] = 0.0f; }
  __syncthreads();

  // sweep 1: row maxima
  for (int c0 = 0; c0 < Lk; c0 += CHK) {
    score_chunk(S, Kb, c0, qbase, aq0, aq1, band, win);
    __syncthreads();
    if (tid < 16) {
      float m = rmax[tid];
      for (int c = 0; c < CHK; ++c) m = fmaxf(m, S[tid][c]);
      rmax[tid] = m;
    }
    __syncthreads();
  }

  // sweep 2: probs + O = P*V
  v8f acc = {};
  for (int c0 = 0; c0 < Lk; c0 += CHK) {
    score_chunk(S, Kb, c0, qbase, aq0, aq1, band, win);
    __syncthreads();
    for (int idx = tid; idx < 16 * CHK; idx += 128) {
      int r = idx >> 8;          // CHK == 256
      int c = idx & (CHK - 1);
      float p = expf(S[r][c] - rmax[r]);
      S[r][c] = p;
      P[r][c] = (bf16_t)p;
    }
    __syncthreads();
    if (tid < 16) {
      float s = rsum[tid];
      for (int c = 0; c < CHK; ++c) s += S[tid][c];
      rsum[tid] = s;
    }
    __syncthreads();
    const bf16_t* Pp = &P[0][0];
    const bf16_t* Vw = Vb + (long)(w * 16) * Lk + c0;
    for (int k0 = 0; k0 < CHK; k0 += 32) {
      v16bf af  = load_frag(Pp + k0, CHK);
      v16bf bfr = load_frag(Vw + k0, Lk);
      acc = WMMA_BF16(af, bfr, acc);
    }
    __syncthreads();
  }

  int n  = lane & 15;
  int dc = w * 16 + n;
  int rg = (lane >> 4) << 3;
#pragma unroll
  for (int j = 0; j < 8; ++j) {
    int m = rg + j;
    float v = acc[j] / rsum[m];
    O[z * osb + (long)(qbase + m) * D_MODEL + h * DKH + dc] = (bf16_t)v;
  }
}

// ---------------------------------------------------------------------------
// Conversions / layout helpers
// ---------------------------------------------------------------------------
__global__ void cvt_f32_bf16(const float* __restrict__ src, bf16_t* __restrict__ dst, long n) {
  long i = (long)blockIdx.x * blockDim.x + threadIdx.x;
  if (i < n) dst[i] = (bf16_t)src[i];
}

__global__ void transpose_w(const float* __restrict__ W, bf16_t* __restrict__ Wt, int K, int N) {
  long i = (long)blockIdx.x * blockDim.x + threadIdx.x;
  if (i < (long)K * N) {
    long k = i / N; int n = (int)(i - k * N);
    Wt[(long)n * K + k] = (bf16_t)W[i];
  }
}

__global__ void build_xsp(const float* __restrict__ x, bf16_t* __restrict__ xsp) {
  long i = (long)blockIdx.x * blockDim.x + threadIdx.x;
  long total = (long)NB * 1024 * D_MODEL;
  if (i < total) {
    int d = (int)(i % D_MODEL); long t = i / D_MODEL;
    int b = (int)(t / 1024); int s = (int)(t % 1024);
    xsp[i] = (bf16_t)x[((long)b * LSEQ + 2 * s) * D_MODEL + d];
  }
}

__global__ void build_vt(const bf16_t* __restrict__ V, bf16_t* __restrict__ Vt, int Lk) {
  long i = (long)blockIdx.x * blockDim.x + threadIdx.x;
  long total = (long)NB * NH * DKH * Lk;
  if (i < total) {
    int key = (int)(i % Lk); long t = i / Lk;
    int d = (int)(t % DKH); t /= DKH;
    int h = (int)(t % NH); int b = (int)(t / NH);
    Vt[i] = V[((long)b * Lk + key) * D_MODEL + h * DKH + d];
  }
}

// ---------------------------------------------------------------------------
// Residual + LayerNorm, one block per token, D = 512
// ---------------------------------------------------------------------------
__global__ __launch_bounds__(256)
void resid_ln(const float* __restrict__ stage, const float* __restrict__ x,
              const float* __restrict__ g, const float* __restrict__ beta,
              float* __restrict__ out)
{
  __shared__ float red[256];
  __shared__ float s_mu, s_rstd;
  long row = blockIdx.x;
  int  t   = threadIdx.x;
  long b0  = row * D_MODEL;

  float v0 = stage[b0 + t]       + x[b0 + t];
  float v1 = stage[b0 + t + 256] + x[b0 + t + 256];
  red[t] = v0 + v1;
  __syncthreads();
  for (int off = 128; off > 0; off >>= 1) {
    if (t < off) red[t] += red[t + off];
    __syncthreads();
  }
  if (t == 0) s_mu = red[0] * (1.0f / 512.0f);
  __syncthreads();
  float mu = s_mu;
  float d0 = v0 - mu, d1 = v1 - mu;
  red[t] = d0 * d0 + d1 * d1;
  __syncthreads();
  for (int off = 128; off > 0; off >>= 1) {
    if (t < off) red[t] += red[t + off];
    __syncthreads();
  }
  if (t == 0) s_rstd = rsqrtf(red[0] * (1.0f / 512.0f) + 1e-5f);
  __syncthreads();
  float rs = s_rstd;
  out[b0 + t]       = d0 * rs * g[t]       + beta[t];
  out[b0 + t + 256] = d1 * rs * g[t + 256] + beta[t + 256];
}

// ---------------------------------------------------------------------------
// Host orchestration
// ---------------------------------------------------------------------------
extern "C" void kernel_launch(void* const* d_in, const int* in_sizes, int n_in,
                              void* d_out, int out_size, void* d_ws, size_t ws_size,
                              hipStream_t stream) {
  (void)in_sizes; (void)n_in; (void)out_size; (void)ws_size;

  const float* xf   = (const float*)d_in[28];
  const float* ln_g = (const float*)d_in[26];
  const float* ln_b = (const float*)d_in[27];

  // workspace carve-out
  char* ws = (char*)d_ws;
  size_t off = 0;
  auto carve = [&](size_t bytes) -> void* {
    void* p = ws + off;
    off += (bytes + 255) & ~(size_t)255;
    return p;
  };

  bf16_t* xb    = (bf16_t*)carve((size_t)NB * LSEQ * D_MODEL * 2);      // x in bf16
  bf16_t* xsp   = (bf16_t*)carve((size_t)NB * 1024 * D_MODEL * 2);      // strided low
  bf16_t* wt[12];
  for (int i = 0; i < 12; ++i) wt[i] = (bf16_t*)carve((size_t)512 * 512 * 2);
  bf16_t* fwt   = (bf16_t*)carve((size_t)512 * 1024 * 2);               // fused W^T
  bf16_t* bufQ  = (bf16_t*)carve((size_t)NB * HALFL * D_MODEL * 2);
  bf16_t* bufK  = (bf16_t*)carve((size_t)NB * HALFL * D_MODEL * 2);
  bf16_t* bufV  = (bf16_t*)carve((size_t)NB * HALFL * D_MODEL * 2);
  bf16_t* bufVt = (bf16_t*)carve((size_t)NB * HALFL * D_MODEL * 2);
  bf16_t* bufA  = (bf16_t*)carve((size_t)NB * HALFL * D_MODEL * 2);     // attn out
  bf16_t* ccat  = (bf16_t*)carve((size_t)NB * HALFL * 1024 * 2);        // [high|cross]
  float*  stage = (float*) carve((size_t)NB * LSEQ * D_MODEL * 4);      // pre-LN fp32

  // conversions
  {
    long n = (long)NB * LSEQ * D_MODEL;
    cvt_f32_bf16<<<(unsigned)((n + 255) / 256), 256, 0, stream>>>(xf, xb, n);
    long ns = (long)NB * 1024 * D_MODEL;
    build_xsp<<<(unsigned)((ns + 255) / 256), 256, 0, stream>>>(xf, xsp);
    for (int i = 0; i < 12; ++i)
      transpose_w<<<(512 * 512 + 255) / 256, 256, 0, stream>>>(
          (const float*)d_in[2 * i], wt[i], 512, 512);
    transpose_w<<<(1024 * 512 + 255) / 256, 256, 0, stream>>>(
        (const float*)d_in[24], fwt, 1024, 512);
  }

  const long SB_FULL = (long)LSEQ  * D_MODEL;   // batch stride of (B,L,D)
  const long SB_HALF = (long)HALFL * D_MODEL;   // batch stride of (B,2048,D)
  const long SB_SP   = (long)1024  * D_MODEL;

  auto gemm = [&](const bf16_t* A, long asb, int lda, const bf16_t* Wt_, int K,
                  const float* bias, float* oF, bf16_t* oB, long osb, int ldo,
                  int ncolOff, int M, int gelu) {
    dim3 g(2, (unsigned)(M / 16), NB);          // N == 512 always -> N/256 = 2
    gemm_bf16<<<g, 128, 0, stream>>>(A, asb, lda, Wt_, K, bias, oF, oB,
                                     osb, ldo, ncolOff, gelu);
  };
  auto attn = [&](const bf16_t* Qp, long qsb, const bf16_t* Kp, long ksb,
                  const bf16_t* Vtp, int Lk, bf16_t* Op, long osb,
                  int band, int win) {
    dim3 g(HALFL / 16, NH, NB);
    attn_kernel<<<g, 128, 0, stream>>>(Qp, qsb, Kp, ksb, Vtp, Lk, Op, osb, band, win);
  };

  const bf16_t* xlow  = xb;
  const bf16_t* xhigh = xb + (long)HALFL * D_MODEL;

  // ---- low branch: q from x_low, k/v from strided x_low ----
  gemm(xlow, SB_FULL, D_MODEL, wt[0], 512, (const float*)d_in[1],
       nullptr, bufQ, SB_HALF, D_MODEL, 0, HALFL, 0);
  gemm(xsp, SB_SP, D_MODEL, wt[1], 512, (const float*)d_in[3],
       nullptr, bufK, SB_SP, D_MODEL, 0, 1024, 0);
  gemm(xsp, SB_SP, D_MODEL, wt[2], 512, (const float*)d_in[5],
       nullptr, bufV, SB_SP, D_MODEL, 0, 1024, 0);
  {
    long n = (long)NB * NH * DKH * 1024;
    build_vt<<<(unsigned)((n + 255) / 256), 256, 0, stream>>>(bufV, bufVt, 1024);
  }
  attn(bufQ, SB_HALF, bufK, SB_SP, bufVt, 1024, bufA, SB_HALF, 0, 0);
  gemm(bufA, SB_HALF, D_MODEL, wt[3], 512, (const float*)d_in[7],
       stage, nullptr, SB_FULL, D_MODEL, 0, HALFL, 0);        // low_out -> rows 0..2047

  // ---- high branch: banded self-attention on x_high ----
  gemm(xhigh, SB_FULL, D_MODEL, wt[4], 512, (const float*)d_in[9],
       nullptr, bufQ, SB_HALF, D_MODEL, 0, HALFL, 0);
  gemm(xhigh, SB_FULL, D_MODEL, wt[5], 512, (const float*)d_in[11],
       nullptr, bufK, SB_HALF, D_MODEL, 0, HALFL, 0);
  gemm(xhigh, SB_FULL, D_MODEL, wt[6], 512, (const float*)d_in[13],
       nullptr, bufV, SB_HALF, D_MODEL, 0, HALFL, 0);
  {
    long n = (long)NB * NH * DKH * HALFL;
    build_vt<<<(unsigned)((n + 255) / 256), 256, 0, stream>>>(bufV, bufVt, HALFL);
  }
  attn(bufQ, SB_HALF, bufK, SB_HALF, bufVt, HALFL, bufA, SB_HALF, 1, 15);
  gemm(bufA, SB_HALF, D_MODEL, wt[7], 512, (const float*)d_in[15],
       nullptr, ccat, (long)HALFL * 1024, 1024, 0, HALFL, 0); // concat[:, :512]

  // ---- cross branch: q from x_high, k/v from x_low ----
  gemm(xhigh, SB_FULL, D_MODEL, wt[8], 512, (const float*)d_in[17],
       nullptr, bufQ, SB_HALF, D_MODEL, 0, HALFL, 0);
  gemm(xlow, SB_FULL, D_MODEL, wt[9], 512, (const float*)d_in[19],
       nullptr, bufK, SB_HALF, D_MODEL, 0, HALFL, 0);
  gemm(xlow, SB_FULL, D_MODEL, wt[10], 512, (const float*)d_in[21],
       nullptr, bufV, SB_HALF, D_MODEL, 0, HALFL, 0);
  {
    long n = (long)NB * NH * DKH * HALFL;
    build_vt<<<(unsigned)((n + 255) / 256), 256, 0, stream>>>(bufV, bufVt, HALFL);
  }
  attn(bufQ, SB_HALF, bufK, SB_HALF, bufVt, HALFL, bufA, SB_HALF, 0, 0);
  gemm(bufA, SB_HALF, D_MODEL, wt[11], 512, (const float*)d_in[23],
       nullptr, ccat, (long)HALFL * 1024, 1024, 512, HALFL, 0); // concat[:, 512:]

  // ---- fused projection + exact GELU -> rows 2048..4095 ----
  gemm(ccat, (long)HALFL * 1024, 1024, fwt, 1024, (const float*)d_in[25],
       stage + (long)HALFL * D_MODEL, nullptr, SB_FULL, D_MODEL, 0, HALFL, 1);

  // ---- residual + LayerNorm ----
  resid_ln<<<NB * LSEQ, 256, 0, stream>>>(stage, xf, ln_g, ln_b, (float*)d_out);
}
